// RWKVEncoder_27049704030487
// MI455X (gfx1250) — matrix operand order repaired and checked by reading, not compiled, not run on hardware
//
#include <hip/hip_runtime.h>
#include <stdint.h>

#define LN_EPS 1e-5f

typedef __attribute__((ext_vector_type(16))) __bf16 v16bf;
typedef __attribute__((ext_vector_type(8)))  float  v8f;
typedef __attribute__((ext_vector_type(4)))  unsigned int u32x4;
typedef __attribute__((ext_vector_type(8)))  unsigned int u32x8;

union Frag16 {
  v16bf v;
  unsigned short u[16];
  unsigned int   w[8];
};

__device__ __forceinline__ unsigned short f2bf(float f) {
  unsigned int u = __float_as_uint(f);
  unsigned int r = u + 0x7FFFu + ((u >> 16) & 1u);   // round-to-nearest-even
  return (unsigned short)(r >> 16);
}
__device__ __forceinline__ float bf2f(unsigned short h) {
  return __uint_as_float(((unsigned int)h) << 16);
}

// ---------------------------------------------------------------------------
// fp32 -> bf16 conversion (weights / input)
// ---------------------------------------------------------------------------
__global__ void cvt_f32_to_bf16(const float* __restrict__ x,
                                unsigned short* __restrict__ y, int n) {
  int i = blockIdx.x * blockDim.x + threadIdx.x;
  if (i < n) y[i] = f2bf(x[i]);
}

// ---------------------------------------------------------------------------
// Generic tiled GEMM: C(MxN) = act(A(MxK) @ Bw(KxN) + bias) [* mul] [+ res]
// A, Bw bf16 row-major.  64x64 tile per workgroup, 8 waves, each wave 32x16.
// B tile (32xK-step x 64 cols) is staged into LDS by the Tensor Data Mover
// (tensor_load_to_lds, ISA ch.8): wave 0 issues the 2D-descriptor DMA and
// waits on TENSORcnt; the workgroup barrier publishes the tile.
// act: 0=none, 1=sigmoid, 2=relu^2
// ---------------------------------------------------------------------------
__global__ __launch_bounds__(256) void gemm_bf16_wmma(
    const unsigned short* __restrict__ A, const unsigned short* __restrict__ Bw,
    const float* bias, const float* mul, const float* res,
    float* out32, unsigned short* out16, int M, int N, int K, int act) {
  __shared__ unsigned short As[64 * 32];
  __shared__ unsigned short Bs[32 * 64];
  const int tid  = threadIdx.x;
  const int lane = tid & 31;
  const int wave = tid >> 5;
  const int m_base = blockIdx.y * 64;
  const int n_base = blockIdx.x * 64;
  const int wm = (wave >> 2) * 32;   // 0 or 32
  const int wn = (wave & 3) * 16;    // 0..48
  const int nn   = lane & 15;
  const int half = lane >> 4;

  v8f acc0 = {}; v8f acc1 = {};

  const int ar = tid >> 2, ac = (tid & 3) * 8;   // A tile: 64 rows x 32 cols

  // Loop-invariant part of the TDM descriptor (D# group 1), ISA 8.4:
  //   [17:16] data_size=1 (2 bytes)   [79:48] tensor_dim0 = N
  //   [111:80] tensor_dim1 = K        [127:112] tile_dim0 = 64
  //   [143:128] tile_dim1 = 32        [207:160] tensor_dim0_stride = N
  const u32x8 g1 = {
      0x00010000u,                                             // wg_mask=0, data_size=2B
      ((unsigned)N & 0xFFFFu) << 16,                           // tensor_dim0 lo
      (((unsigned)N >> 16) & 0xFFFFu) | (((unsigned)K & 0xFFFFu) << 16),
      (((unsigned)K >> 16) & 0xFFFFu) | (64u << 16),           // tile_dim0 = 64
      32u,                                                     // tile_dim1 = 32
      (unsigned)N,                                             // dim0 stride lo
      0u, 0u };
  const unsigned lds_b = (unsigned)(uintptr_t)(&Bs[0]);

  for (int k0 = 0; k0 < K; k0 += 32) {
    const unsigned short* ga = A + (size_t)(m_base + ar) * K + (k0 + ac);
    *(uint4*)&As[ar * 32 + ac] = *(const uint4*)ga;
    if (k0 + 32 < K) __builtin_prefetch(ga + 32, 0, 0);        // global_prefetch_b8

    if (wave == 0) {
      // D# group 0: count=1 | lds_addr | global tile address | type=2
      unsigned long long gaddr =
          (unsigned long long)(uintptr_t)(Bw + (size_t)k0 * N + n_base);
      u32x4 g0 = { 1u, lds_b,
                   (unsigned)(gaddr & 0xFFFFFFFFull),
                   (unsigned)((gaddr >> 32) & 0x1FFFFFFull) | (2u << 30) };
      asm volatile("tensor_load_to_lds %0, %1" : : "s"(g0), "s"(g1) : "memory");
      __builtin_amdgcn_s_wait_tensorcnt(0);
    }
    __syncthreads();

    // B fragment (32x16): lane = N column, half-wave splits K (ISA 7.12.2)
    Frag16 bf;
#pragma unroll
    for (int v = 0; v < 8; ++v) {
      int kk = ((v < 4) ? 2 * v : 16 + 2 * (v - 4)) + 8 * half;
      bf.u[2 * v]     = Bs[kk * 64 + wn + nn];
      bf.u[2 * v + 1] = Bs[(kk + 1) * 64 + wn + nn];
    }
    // A fragments (16x32): lane = M row; K pairs per documented layout
    Frag16 a0, a1;
#pragma unroll
    for (int v = 0; v < 8; ++v) {
      int kk = ((v < 4) ? 2 * v : 16 + 2 * (v - 4)) + (half ? 8 : 0);
      a0.w[v] = *(const unsigned int*)&As[(wm + nn) * 32 + kk];
      a1.w[v] = *(const unsigned int*)&As[(wm + 16 + nn) * 32 + kk];
    }
    acc0 = __builtin_amdgcn_wmma_f32_16x16x32_bf16(false, a0.v, false, bf.v,
                                                   (short)0, acc0, false, false);
    acc1 = __builtin_amdgcn_wmma_f32_16x16x32_bf16(false, a1.v, false, bf.v,
                                                   (short)0, acc1, false, false);
    __syncthreads();
  }

  const int ncol = n_base + wn + nn;
  const float bval = bias ? bias[ncol] : 0.0f;
#pragma unroll
  for (int v = 0; v < 8; ++v) {
    int   rows[2] = { m_base + wm + v + 8 * half, m_base + wm + 16 + v + 8 * half };
    float vals[2] = { acc0[v], acc1[v] };
#pragma unroll
    for (int g = 0; g < 2; ++g) {
      float x = vals[g] + bval;
      if (act == 1)      x = 1.0f / (1.0f + __expf(-x));
      else if (act == 2) { x = fmaxf(x, 0.0f); x = x * x; }
      size_t idx = (size_t)rows[g] * N + ncol;
      if (mul)   x *= mul[idx];
      if (res)   x += res[idx];
      if (out32) out32[idx] = x;
      if (out16) out16[idx] = f2bf(x);
    }
  }
}

// ---------------------------------------------------------------------------
// Conv (3x3, pad 1) as implicit GEMM on WMMA.
// A = weights (Cout x Cin*9) bf16 row-major, B = im2col patches gathered on
// the fly.  One 16x16 output tile per wave.  Epilogue: +bias, relu, bf16 out.
// embed_layout=1 writes (b, u, co*Wout + f) for the embed GEMM A-matrix.
// ---------------------------------------------------------------------------
__global__ __launch_bounds__(256) void conv_gemm_wmma(
    const unsigned short* __restrict__ In, const unsigned short* __restrict__ Wt,
    const float* __restrict__ bias, unsigned short* __restrict__ Out,
    int Cin, int Hin, int Win, int Cout, int Hout, int Wout,
    int stride, int Ktot, int embed_layout) {
  const int tid  = threadIdx.x;
  const int lane = tid & 31;
  const int wave = tid >> 5;
  const int nn   = lane & 15;
  const int half = lane >> 4;
  const int Mt   = Cout >> 4;
  const int tile = blockIdx.x * 8 + wave;
  const int mt = tile % Mt;
  const int nt = tile / Mt;
  const int ncol = nt * 16 + nn;
  const int HW  = Hout * Wout;
  const int b   = ncol / HW;
  const int rem = ncol - b * HW;
  const int u   = rem / Wout;
  const int f   = rem - u * Wout;
  const int h0  = u * stride - 1;
  const int w0  = f * stride - 1;
  const int co_a = mt * 16 + nn;

  v8f acc = {};
  for (int k0 = 0; k0 < Ktot; k0 += 32) {
    Frag16 af;
#pragma unroll
    for (int v = 0; v < 8; ++v) {
      int kk = ((v < 4) ? 2 * v : 16 + 2 * (v - 4)) + (half ? 8 : 0) + k0;
      af.u[2 * v]     = (kk     < Ktot) ? Wt[(size_t)co_a * Ktot + kk]     : (unsigned short)0;
      af.u[2 * v + 1] = (kk + 1 < Ktot) ? Wt[(size_t)co_a * Ktot + kk + 1] : (unsigned short)0;
    }
    Frag16 bf;
#pragma unroll
    for (int v = 0; v < 8; ++v) {
#pragma unroll
      for (int j = 0; j < 2; ++j) {
        int k = k0 + ((v < 4) ? 2 * v : 16 + 2 * (v - 4)) + j + 8 * half;
        unsigned short val = 0;
        if (k < Ktot) {
          int ci = k / 9;
          int r9 = k - ci * 9;
          int kh = r9 / 3;
          int kw = r9 - kh * 3;
          int hi = h0 + kh;
          int wi = w0 + kw;
          if (hi >= 0 && hi < Hin && wi >= 0 && wi < Win)
            val = In[(((size_t)b * Cin + ci) * Hin + hi) * Win + wi];
        }
        bf.u[2 * v + j] = val;
      }
    }
    acc = __builtin_amdgcn_wmma_f32_16x16x32_bf16(false, af.v, false, bf.v,
                                                  (short)0, acc, false, false);
  }
#pragma unroll
  for (int v = 0; v < 8; ++v) {
    int co = mt * 16 + v + 8 * half;
    float x = fmaxf(acc[v] + bias[co], 0.0f);
    size_t idx;
    if (embed_layout)
      idx = ((size_t)b * Hout + u) * ((size_t)Cout * Wout) + (size_t)co * Wout + f;
    else
      idx = (((size_t)b * Cout + co) * Hout + u) * Wout + f;
    Out[idx] = f2bf(x);
  }
}

// ---------------------------------------------------------------------------
// LayerNorm: one wave per row (wave32 shuffle reduction over D=512)
// ---------------------------------------------------------------------------
__global__ __launch_bounds__(256) void ln_wave(
    const float* __restrict__ in, const float* __restrict__ g,
    const float* __restrict__ b, float* out32, unsigned short* out16, int D) {
  const int lane = threadIdx.x & 31;
  const int wave = threadIdx.x >> 5;
  const size_t row = (size_t)blockIdx.x * 8 + wave;
  const float* x = in + row * D;
  float s = 0.f, s2 = 0.f;
  for (int i = lane; i < D; i += 32) { float v = x[i]; s += v; s2 += v * v; }
  for (int o = 16; o > 0; o >>= 1) {
    s  += __shfl_xor(s,  o, 32);
    s2 += __shfl_xor(s2, o, 32);
  }
  const float mean = s / D;
  const float var  = s2 / D - mean * mean;
  const float rstd = rsqrtf(var + LN_EPS);
  for (int i = lane; i < D; i += 32) {
    float y = (x[i] - mean) * rstd * g[i] + b[i];
    if (out32) out32[row * D + i] = y;
    if (out16) out16[row * D + i] = f2bf(y);
  }
}

// ---------------------------------------------------------------------------
// Time-shift mixing: out = bf16(xn[t]*mix + xn[t-1]*(1-mix)) for up to 3 mixes
// ---------------------------------------------------------------------------
__global__ void mix_shift(const float* __restrict__ xn,
                          const float* __restrict__ mk, const float* __restrict__ mv,
                          const float* __restrict__ mr,
                          unsigned short* ok, unsigned short* ov, unsigned short* orr,
                          int U, int D, int total) {
  int i = blockIdx.x * blockDim.x + threadIdx.x;
  if (i >= total) return;
  int d   = i % D;
  int row = i / D;
  int u   = row % U;
  float cur  = xn[i];
  float prev = (u == 0) ? 0.0f : xn[i - D];
  if (ok)  { float m = mk[d]; ok[i]  = f2bf(cur * m + prev * (1.f - m)); }
  if (ov)  { float m = mv[d]; ov[i]  = f2bf(cur * m + prev * (1.f - m)); }
  if (orr) { float m = mr[d]; orr[i] = f2bf(cur * m + prev * (1.f - m)); }
}

// ---------------------------------------------------------------------------
// Numerically stable RWKV-4 WKV scan: one thread per (b, d)
// ---------------------------------------------------------------------------
__global__ void wkv_scan(const float* __restrict__ decay, const float* __restrict__ first,
                         const float* __restrict__ k, const float* __restrict__ v,
                         float* __restrict__ out, int Bb, int U, int D) {
  int t = blockIdx.x * blockDim.x + threadIdx.x;
  if (t >= Bb * D) return;
  int b = t / D, d = t % D;
  float w  = -__expf(decay[d]);
  float uu = first[d];
  float aa = 0.f, bb = 0.f, pp = -1e38f;
  size_t base = (size_t)b * U * D + d;
  for (int i = 0; i < U; ++i) {
    size_t idx = base + (size_t)i * D;
    float kt = k[idx], vt = v[idx];
    float ww = uu + kt;
    float p  = fmaxf(pp, ww);
    float e1 = __expf(pp - p), e2 = __expf(ww - p);
    out[idx] = (e1 * aa + e2 * vt) / (e1 * bb + e2);
    float ww2 = pp + w;
    float p2  = fmaxf(ww2, kt);
    e1 = __expf(ww2 - p2); e2 = __expf(kt - p2);
    aa = e1 * aa + e2 * vt;
    bb = e1 * bb + e2;
    pp = p2;
  }
}

// out = bf16(r * wkv)
__global__ void rmul_bf16(const float* __restrict__ r, const float* __restrict__ wkv,
                          unsigned short* __restrict__ out, int n) {
  int i = blockIdx.x * blockDim.x + threadIdx.x;
  if (i < n) out[i] = f2bf(r[i] * wkv[i]);
}

__global__ void olens_kernel(const int* __restrict__ x_len, float* __restrict__ out, int Bb) {
  int i = threadIdx.x;
  if (i < Bb) {
    int l1 = (x_len[i] - 1) / 2 + 1;
    int o  = (l1 - 1) / 2 + 1;
    out[i] = (float)o;
  }
}

// ---------------------------------------------------------------------------
// Host orchestration
// ---------------------------------------------------------------------------
extern "C" void kernel_launch(void* const* d_in, const int* in_sizes, int n_in,
                              void* d_out, int out_size, void* d_ws, size_t ws_size,
                              hipStream_t stream) {
  (void)in_sizes; (void)n_in; (void)out_size; (void)ws_size;
  constexpr int Bb = 8, L = 4096, F = 80, D = 512, Hh = 2048, NBlk = 4, U = 1024;
  constexpr int C1 = 128, C2 = 256, C3 = 512;
  constexpr int H1 = 2048, W1 = 40, H2 = 1024, W2 = 20;
  constexpr int BU = Bb * U;                  // 8192 rows
  constexpr int Kemb = C3 * W2;               // 10240

  typedef unsigned short bh;
  char* ws = (char*)d_ws;
  size_t off = 0;
  auto alloc = [&](size_t bytes) -> void* {
    void* p = ws + off;
    off += (bytes + 255) & ~(size_t)255;
    return p;
  };

  // bf16 weight copies
  bh* wc1  = (bh*)alloc((size_t)C1 * 9 * 2 + 256);
  bh* wc2  = (bh*)alloc((size_t)C2 * C1 * 9 * 2);
  bh* wc3  = (bh*)alloc((size_t)C3 * C2 * 9 * 2);
  bh* wemb = (bh*)alloc((size_t)Kemb * D * 2);
  bh* wka  = (bh*)alloc((size_t)NBlk * D * D * 2);
  bh* wva  = (bh*)alloc((size_t)NBlk * D * D * 2);
  bh* wra  = (bh*)alloc((size_t)NBlk * D * D * 2);
  bh* woa  = (bh*)alloc((size_t)NBlk * D * D * 2);
  bh* fwk  = (bh*)alloc((size_t)NBlk * D * Hh * 2);
  bh* fwv  = (bh*)alloc((size_t)NBlk * Hh * D * 2);
  bh* fwr  = (bh*)alloc((size_t)NBlk * D * D * 2);
  // activations
  bh* xbf  = (bh*)alloc((size_t)Bb * L * F * 2);
  bh* y1   = (bh*)alloc((size_t)Bb * C1 * H1 * W1 * 2);   // reused as Aemb
  bh* y2   = (bh*)alloc((size_t)Bb * C2 * H2 * W2 * 2);
  bh* Aemb = y1;                                          // same elem count (83.9M)
  float* hbuf  = (float*)alloc((size_t)BU * D * 4);
  float* xnbuf = (float*)alloc((size_t)BU * D * 4);
  float* t0    = (float*)alloc((size_t)BU * D * 4);
  float* kb    = (float*)alloc((size_t)BU * D * 4);
  float* vb    = (float*)alloc((size_t)BU * D * 4);
  float* rb    = (float*)alloc((size_t)BU * D * 4);
  float* wkvb  = (float*)alloc((size_t)BU * D * 4);
  bh* bk  = (bh*)alloc((size_t)BU * D * 2);
  bh* bv  = (bh*)alloc((size_t)BU * D * 2);
  bh* br  = (bh*)alloc((size_t)BU * D * 2);
  bh* kkb = (bh*)alloc((size_t)BU * Hh * 2);

  auto cvt = [&](const void* src, bh* dst, size_t n) {
    cvt_f32_to_bf16<<<(unsigned)((n + 255) / 256), 256, 0, stream>>>(
        (const float*)src, dst, (int)n);
  };
  cvt(d_in[2],  wc1,  (size_t)C1 * 9);
  cvt(d_in[4],  wc2,  (size_t)C2 * C1 * 9);
  cvt(d_in[6],  wc3,  (size_t)C3 * C2 * 9);
  cvt(d_in[8],  wemb, (size_t)Kemb * D);
  cvt(d_in[19], wka,  (size_t)NBlk * D * D);
  cvt(d_in[20], wva,  (size_t)NBlk * D * D);
  cvt(d_in[21], wra,  (size_t)NBlk * D * D);
  cvt(d_in[22], woa,  (size_t)NBlk * D * D);
  cvt(d_in[27], fwk,  (size_t)NBlk * D * Hh);
  cvt(d_in[28], fwv,  (size_t)NBlk * Hh * D);
  cvt(d_in[29], fwr,  (size_t)NBlk * D * D);
  cvt(d_in[0],  xbf,  (size_t)Bb * L * F);

  // conv frontend (implicit-GEMM WMMA)
  {
    int N = Bb * H1 * W1;
    conv_gemm_wmma<<<(C1 / 16) * (N / 16) / 8, 256, 0, stream>>>(
        xbf, wc1, (const float*)d_in[3], y1, 1, L, F, C1, H1, W1, 2, 9, 0);
  }
  {
    int N = Bb * H2 * W2;
    conv_gemm_wmma<<<(C2 / 16) * (N / 16) / 8, 256, 0, stream>>>(
        y1, wc2, (const float*)d_in[5], y2, C1, H1, W1, C2, H2, W2, 2, C1 * 9, 0);
  }
  {
    int N = Bb * H2 * W2;   // conv3 output -> embed A-matrix layout (aliases y1)
    conv_gemm_wmma<<<(C3 / 16) * (N / 16) / 8, 256, 0, stream>>>(
        y2, wc3, (const float*)d_in[7], Aemb, C2, H2, W2, C3, H2, W2, 1, C2 * 9, 1);
  }

  auto gemm = [&](const bh* A, const bh* Bw, const float* bias, const float* mul,
                  const float* res, float* o32, bh* o16, int M, int N, int K, int act) {
    dim3 g(N / 64, M / 64);
    gemm_bf16_wmma<<<g, 256, 0, stream>>>(A, Bw, bias, mul, res, o32, o16, M, N, K, act);
  };

  // embed + LN
  gemm(Aemb, wemb, (const float*)d_in[9], nullptr, nullptr, t0, nullptr, BU, D, Kemb, 0);
  ln_wave<<<BU / 8, 256, 0, stream>>>(t0, (const float*)d_in[10], (const float*)d_in[11],
                                      hbuf, nullptr, D);

  const int nEl = BU * D;
  for (int i = 0; i < NBlk; ++i) {
    const size_t oD = (size_t)i * D;
    // --- time mixing ---
    ln_wave<<<BU / 8, 256, 0, stream>>>(hbuf, (const float*)d_in[12] + oD,
                                        (const float*)d_in[13] + oD, xnbuf, nullptr, D);
    mix_shift<<<(nEl + 255) / 256, 256, 0, stream>>>(
        xnbuf, (const float*)d_in[16] + oD, (const float*)d_in[17] + oD,
        (const float*)d_in[18] + oD, bk, bv, br, U, D, nEl);
    gemm(bk, wka + (size_t)i * D * D, nullptr, nullptr, nullptr, kb, nullptr, BU, D, D, 0);
    gemm(bv, wva + (size_t)i * D * D, nullptr, nullptr, nullptr, vb, nullptr, BU, D, D, 0);
    gemm(br, wra + (size_t)i * D * D, nullptr, nullptr, nullptr, rb, nullptr, BU, D, D, 1);
    wkv_scan<<<(Bb * D + 255) / 256, 256, 0, stream>>>(
        (const float*)d_in[14] + oD, (const float*)d_in[15] + oD, kb, vb, wkvb, Bb, U, D);
    rmul_bf16<<<(nEl + 255) / 256, 256, 0, stream>>>(rb, wkvb, bk, nEl);
    gemm(bk, woa + (size_t)i * D * D, nullptr, nullptr, hbuf, hbuf, nullptr, BU, D, D, 0);
    // --- channel mixing ---
    ln_wave<<<BU / 8, 256, 0, stream>>>(hbuf, (const float*)d_in[23] + oD,
                                        (const float*)d_in[24] + oD, xnbuf, nullptr, D);
    mix_shift<<<(nEl + 255) / 256, 256, 0, stream>>>(
        xnbuf, (const float*)d_in[25] + oD, nullptr,
        (const float*)d_in[26] + oD, bk, nullptr, br, U, D, nEl);
    gemm(bk, fwk + (size_t)i * D * Hh, nullptr, nullptr, nullptr, nullptr, kkb, BU, Hh, D, 2);
    gemm(br, fwr + (size_t)i * D * D, nullptr, nullptr, nullptr, rb, nullptr, BU, D, D, 1);
    gemm(kkb, fwv + (size_t)i * Hh * D, nullptr, rb, hbuf, hbuf, nullptr, BU, D, Hh, 0);
  }

  // final LN -> d_out, then olens
  ln_wave<<<BU / 8, 256, 0, stream>>>(hbuf, (const float*)d_in[30], (const float*)d_in[31],
                                      (float*)d_out, nullptr, D);
  olens_kernel<<<1, 32, 0, stream>>>((const int*)d_in[1], (float*)d_out + (size_t)BU * D, Bb);
}
